// GraphAttentionEmbeddingLayer_40802189312768
// MI455X (gfx1250) — compile-verified, matrix operand order checked by reading, not run on hardware
//
#include <hip/hip_runtime.h>
#include <hip/hip_bf16.h>
#include <cstdint>
#include <cstddef>

typedef __attribute__((ext_vector_type(16))) _Float16 v16h;
typedef __attribute__((ext_vector_type(8)))  float    v8f;

#define BB     8
#define NN     2048
#define BN     (BB*NN)
#define D_IN_  64
#define F_IN_  128
#define F_OUT_ 64
#define EDIM   128
#define FDIM   128
#define LSEQ   50
#define CONC   384
#define TXTOUT 64

// ---------------------------------------------------------------------------
// Workspace layout (bytes)
//  wbigF  : f16 B-fragments of [128 x 1536] concat conv weights   393216
//  wF     : f16 B-fragments of [128 x 64]  projection W            16384
//  textOut: f32 [16384 x 64]                                     4194304
//  hBuf   : f32 [16384 x 64]                                     4194304
//  hfrag  : f16 B-fragments of h per batch (attention GEMM)      2097152
//  asrc   : f32 [16384]                                            65536
//  adst   : f32 [16384]                                            65536
// ---------------------------------------------------------------------------
#define OFF_WBIGF  0
#define OFF_WF     393216
#define OFF_TEXT   409600
#define OFF_H      4603904
#define OFF_HFRAG  8798208
#define OFF_ASRC   10895360
#define OFF_ADST   10960896

// Fragment layout conventions (A/B use this file's contiguous K order; C as in the ISA):
//  A (16x32 f16): lane l holds row (l&15); halves j -> K = kstep*32 + ((l>>4)<<4) + j
//  B (32x16 f16): lane l holds col (l&15); halves j -> K = kstep*32 + ((l>>4)<<4) + j
//  C (16x16 f32): vgpr v, lane l -> M = (l<16 ? v : v+8), N = (l&15)
// B fragments stored flat: frag(ntile,kstep) = 32 lanes x 16 halves = 512 halves.

// ---------------------------------------------------------------------------
// Kernel 0: pre-swizzle weights into B-fragment order (f32 -> f16)
// ---------------------------------------------------------------------------
__global__ void prep_weights_kernel(const float* __restrict__ w3,
                                    const float* __restrict__ w4,
                                    const float* __restrict__ w5,
                                    const float* __restrict__ W,
                                    _Float16* __restrict__ wbigF,
                                    _Float16* __restrict__ wF) {
  int idx = blockIdx.x * 256 + threadIdx.x;
  if (idx < 96 * 4 * 512) {              // Wbig: 96 ntiles x 4 ksteps x 512 halves
    int j = idx & 15, lane = (idx >> 4) & 31, s = (idx >> 9) & 3, ntG = idx >> 11;
    int e   = s * 32 + ((lane >> 4) << 4) + j;   // K index = embedding dim
    int col = ntG * 16 + (lane & 15);            // concat conv output column
    const float* w; int tap, f;
    if (col < 384)      { w = w3; tap = col >> 7;          f = col & 127; }
    else if (col < 896) { w = w4; tap = (col - 384) >> 7;  f = (col - 384) & 127; }
    else                { w = w5; tap = (col - 896) >> 7;  f = (col - 896) & 127; }
    wbigF[idx] = (_Float16)w[((size_t)tap * EDIM + e) * FDIM + f];
  }
  int idx2 = idx - 96 * 4 * 512;
  if (idx2 >= 0 && idx2 < 4 * 4 * 512) { // W: 4 ntiles x 4 ksteps x 512 halves
    int j = idx2 & 15, lane = (idx2 >> 4) & 31, s = (idx2 >> 9) & 3, nt = idx2 >> 11;
    int k = s * 32 + ((lane >> 4) << 4) + j;
    int n = nt * 16 + (lane & 15);
    wF[idx2] = (_Float16)W[(size_t)k * F_OUT_ + n];
  }
}

// ---------------------------------------------------------------------------
// Kernel 1: TextCNN — one sentence per workgroup (8 waves).
//  emb(50x128,f16 in LDS) @ Wbig(128x1536) via WMMA, per-filter passes,
//  shift-combine + ReLU + maxpool in LDS, then 384->64 linear.
// ---------------------------------------------------------------------------
__global__ __launch_bounds__(256)
void textcnn_kernel(const int*   __restrict__ text_inp,
                    const float* __restrict__ embed,
                    const float* __restrict__ b3,
                    const float* __restrict__ b4,
                    const float* __restrict__ b5,
                    const _Float16* __restrict__ wbigF,
                    const float* __restrict__ lin_W,
                    const float* __restrict__ lin_b,
                    float* __restrict__ text_out) {
  __shared__ __attribute__((aligned(32))) _Float16 sEmb[64 * 128];  // 16 KB
  __shared__ __attribute__((aligned(32))) _Float16 sP[64 * 640];    // 80 KB (max pass)
  __shared__ float sPool[CONC];
  __shared__ int   sTok[64];

  const int tid  = threadIdx.x;
  const int lane = tid & 31;
  const int wave = tid >> 5;
  const int sent = blockIdx.x;

  if (tid < 64) sTok[tid] = (tid < LSEQ) ? text_inp[(size_t)sent * LSEQ + tid] : 0;
  __syncthreads();

  // Gather token embeddings -> LDS (rows >= LSEQ zero-padded)
  for (int idx = tid; idx < 64 * 128; idx += 256) {
    int r = idx >> 7, e = idx & 127;
    float v = (r < LSEQ) ? embed[(size_t)sTok[r] * EDIM + e] : 0.f;
    sEmb[idx] = (_Float16)v;
  }
  __syncthreads();

  int colOff16 = 0;  // ntile offset into Wbig of this filter group
  for (int hi = 0; hi < 3; ++hi) {
    const int taps   = 3 + hi;
    const int width  = taps * FDIM;     // 384 / 512 / 640
    const int ntiles = width >> 4;      // 24 / 32 / 40 (all divisible by 8 waves)

    // ---- GEMM pass: P[64 x width] = emb @ Wbig[:, group] ----
    for (int nt = wave; nt < ntiles; nt += 8) {
      const int fragNt = colOff16 + nt;
      for (int mt = 0; mt < 4; ++mt) {
        v8f acc = {};
        const _Float16* aB = &sEmb[(mt * 16 + (lane & 15)) * 128 + ((lane >> 4) << 4)];
        #pragma unroll
        for (int s = 0; s < 4; ++s) {
          v16h a = *(const v16h*)(aB + s * 32);
          v16h b = *(const v16h*)(wbigF + ((size_t)(fragNt * 4 + s) * 32 + lane) * 16);
          acc = __builtin_amdgcn_wmma_f32_16x16x32_f16(false, a, false, b,
                                                       (short)0, acc, false, false);
        }
        const int col   = nt * 16 + (lane & 15);
        const int rbase = mt * 16 + ((lane < 16) ? 0 : 8);
        #pragma unroll
        for (int v = 0; v < 8; ++v)
          sP[(rbase + v) * width + col] = (_Float16)acc[v];
      }
    }
    __syncthreads();

    // ---- shift-combine + bias + ReLU + maxpool ----
    if (tid < FDIM) {
      const int f = tid;
      const float* bptr = (hi == 0) ? b3 : ((hi == 1) ? b4 : b5);
      const float bv = bptr[f];
      const int T = LSEQ - taps + 1;
      float m = 0.f;  // ReLU output is >= 0
      for (int t = 0; t < T; ++t) {
        float c = bv;
        for (int k = 0; k < taps; ++k)
          c += (float)sP[(t + k) * width + k * FDIM + f];
        c = c > 0.f ? c : 0.f;
        m = c > m ? c : m;
      }
      sPool[hi * FDIM + f] = m;
    }
    __syncthreads();   // protect sP before next pass overwrites it
    colOff16 += ntiles;
  }

  // ---- final linear 384 -> 64 ----
  if (tid < TXTOUT) {
    float acc = lin_b[tid];
    for (int i = 0; i < CONC; ++i)
      acc += sPool[i] * lin_W[(size_t)i * TXTOUT + tid];
    text_out[(size_t)sent * TXTOUT + tid] = acc;
  }
}

// ---------------------------------------------------------------------------
// Kernel 2: h = concat([inp, text_out]) @ W   (16384x128 @ 128x64) via WMMA
// ---------------------------------------------------------------------------
__global__ __launch_bounds__(256)
void hproj_kernel(const float* __restrict__ inp,
                  const float* __restrict__ text_out,
                  const _Float16* __restrict__ wF,
                  float* __restrict__ hOut) {
  __shared__ __attribute__((aligned(32))) _Float16 sX[128 * 128];  // 32 KB
  const int tid = threadIdx.x, lane = tid & 31, wave = tid >> 5;
  const int r0 = blockIdx.x * 128;

  for (int idx = tid; idx < 128 * 128; idx += 256) {
    int r = idx >> 7, c = idx & 127;
    float v = (c < D_IN_) ? inp[(size_t)(r0 + r) * D_IN_ + c]
                          : text_out[(size_t)(r0 + r) * TXTOUT + (c - D_IN_)];
    sX[idx] = (_Float16)v;
  }
  __syncthreads();

  v8f acc[4] = {};
  const _Float16* aB = &sX[(wave * 16 + (lane & 15)) * 128 + ((lane >> 4) << 4)];
  #pragma unroll
  for (int s = 0; s < 4; ++s) {
    v16h a = *(const v16h*)(aB + s * 32);
    #pragma unroll
    for (int nt = 0; nt < 4; ++nt) {
      v16h b = *(const v16h*)(wF + ((size_t)(nt * 4 + s) * 32 + lane) * 16);
      acc[nt] = __builtin_amdgcn_wmma_f32_16x16x32_f16(false, a, false, b,
                                                       (short)0, acc[nt], false, false);
    }
  }
  const int rbase = r0 + wave * 16 + ((lane < 16) ? 0 : 8);
  #pragma unroll
  for (int nt = 0; nt < 4; ++nt)
    #pragma unroll
    for (int v = 0; v < 8; ++v)
      hOut[(size_t)(rbase + v) * F_OUT_ + nt * 16 + (lane & 15)] = acc[nt][v];
}

// ---------------------------------------------------------------------------
// Kernel 2b: repack h into f16 B-fragments for the attention GEMM.
// Layout: hfrag[(((b*64 + jc)*4 + nt)*32 + lane)*16 + j],
//         K = jc*32 + ((lane>>4)<<4) + j (node j), N = nt*16 + (lane&15) (feature)
// ---------------------------------------------------------------------------
__global__ void hfrag_kernel(const float* __restrict__ h, _Float16* __restrict__ hfrag) {
  int idx = blockIdx.x * 256 + threadIdx.x;            // 1,048,576 total
  int j = idx & 15, lane = (idx >> 4) & 31, nt = (idx >> 9) & 3;
  int jc = (idx >> 11) & 63, b = idx >> 17;
  int jj = jc * 32 + ((lane >> 4) << 4) + j;
  int f  = nt * 16 + (lane & 15);
  hfrag[idx] = (_Float16)h[(((size_t)b * NN) + jj) * F_OUT_ + f];
}

// ---------------------------------------------------------------------------
// Kernel 2c: a_src = h . a[:64], a_dst = h . a[64:]
// ---------------------------------------------------------------------------
__global__ void attvec_kernel(const float* __restrict__ h,
                              const float* __restrict__ a_vec,
                              float* __restrict__ asrc, float* __restrict__ adst) {
  int r = blockIdx.x * 256 + threadIdx.x;
  if (r >= BN) return;
  float s0 = 0.f, s1 = 0.f;
  for (int f = 0; f < F_OUT_; ++f) {
    float hv = h[(size_t)r * F_OUT_ + f];
    s0 += hv * a_vec[f];
    s1 += hv * a_vec[F_OUT_ + f];
  }
  asrc[r] = s0;
  adst[r] = s1;
}

// ---------------------------------------------------------------------------
// Kernel 3: masked-softmax attention + (attention @ h) + ELU.
// One workgroup per (batch, 16-row tile). Pass 1: online max/sum over the
// adjacency row. Pass 2: rebuild probabilities directly into WMMA A-fragments,
// accumulate 16x64 output split K-wise across 8 waves, reduce in LDS.
// ---------------------------------------------------------------------------
__global__ __launch_bounds__(256)
void attn_kernel(const float* __restrict__ adj,
                 const float* __restrict__ asrc,
                 const float* __restrict__ adst,
                 const _Float16* __restrict__ hfrag,
                 float* __restrict__ out) {
  __shared__ float sAcc[8][1024];   // 32 KB partial C
  __shared__ float sM[16][16], sS[16][16];
  __shared__ float sMax[16], sInv[16], sAs[16];

  const int tid = threadIdx.x, lane = tid & 31, wave = tid >> 5;
  const int b = blockIdx.x >> 7, it = blockIdx.x & 127, i0 = it * 16;

  if (tid < 16) sAs[tid] = asrc[(size_t)b * NN + i0 + tid];
  __syncthreads();

  // ---- pass 1: online softmax stats (16 sub-scanners per row) ----
  {
    const int row = tid >> 4, sub = tid & 15;
    const float* adjRow = adj + (((size_t)b * NN) + i0 + row) * NN;
    const float* adstB  = adst + (size_t)b * NN;
    const float ar = sAs[row];
    float m = -3.0e38f, s = 0.f;
    const int j0 = sub * 128;
    __builtin_prefetch(adjRow + j0, 0, 1);  // global_prefetch_b8
    for (int j = j0; j < j0 + 128; ++j) {
      float av = adjRow[j];
      if (av > 0.f) {
        float x = ar + adstB[j];
        x = x > 0.f ? x : 0.2f * x;          // LeakyReLU(0.2)
        if (x > m) { s = s * __expf(m - x) + 1.f; m = x; }
        else       { s += __expf(x - m); }
      }
    }
    sM[row][sub] = m; sS[row][sub] = s;
  }
  __syncthreads();
  if (tid < 16) {
    float m = -3.0e38f, s = 0.f;
    for (int k = 0; k < 16; ++k) {
      float mk = sM[tid][k], sk = sS[tid][k];
      if (mk > m) { s = s * __expf(m - mk) + sk; m = mk; }
      else        { s += sk * __expf(mk - m); }
    }
    sMax[tid] = m;
    sInv[tid] = 1.f / s;
  }
  __syncthreads();

  // ---- pass 2: P-fragments -> WMMA, K split across waves ----
  v8f acc[4] = {};
  {
    const int row = lane & 15;
    const float mrow = sMax[row], inv = sInv[row], ar = sAs[row];
    const float* adjRow = adj + (((size_t)b * NN) + i0 + row) * NN;
    const float* adstB  = adst + (size_t)b * NN;
    for (int c8 = 0; c8 < 8; ++c8) {
      const int jc = wave * 8 + c8;
      const int jb = jc * 32 + ((lane >> 4) << 4);
      v16h a = {};
      #pragma unroll
      for (int j = 0; j < 16; ++j) {
        const int jj = jb + j;
        float av = adjRow[jj];
        float p = 0.f;
        if (av > 0.f) {
          float x = ar + adstB[jj];
          x = x > 0.f ? x : 0.2f * x;
          p = __expf(x - mrow) * inv;
        }
        a[j] = (_Float16)p;
      }
      const _Float16* hb = hfrag + (((size_t)b * 64 + jc) * 4) * 512;
      #pragma unroll
      for (int nt = 0; nt < 4; ++nt) {
        v16h bf = *(const v16h*)(hb + (size_t)nt * 512 + lane * 16);
        acc[nt] = __builtin_amdgcn_wmma_f32_16x16x32_f16(false, a, false, bf,
                                                         (short)0, acc[nt], false, false);
      }
    }
  }
  {
    const int rb = (lane < 16) ? 0 : 8;
    #pragma unroll
    for (int nt = 0; nt < 4; ++nt)
      #pragma unroll
      for (int v = 0; v < 8; ++v)
        sAcc[wave][(rb + v) * 64 + nt * 16 + (lane & 15)] = acc[nt][v];
  }
  __syncthreads();

  // ---- cross-wave reduce + ELU + store ----
  for (int idx = tid; idx < 1024; idx += 256) {
    float s = 0.f;
    #pragma unroll
    for (int w = 0; w < 8; ++w) s += sAcc[w][idx];
    float r = s > 0.f ? s : (__expf(s) - 1.f);
    out[(((size_t)b * NN) + i0 + (idx >> 6)) * F_OUT_ + (idx & 63)] = r;
  }
}

// ---------------------------------------------------------------------------
extern "C" void kernel_launch(void* const* d_in, const int* in_sizes, int n_in,
                              void* d_out, int out_size, void* d_ws, size_t ws_size,
                              hipStream_t stream) {
  (void)in_sizes; (void)n_in; (void)out_size; (void)ws_size;
  const float* inp   = (const float*)d_in[0];
  const float* adj   = (const float*)d_in[1];
  const int*   text  = (const int*)  d_in[2];
  const float* W     = (const float*)d_in[3];
  const float* a_vec = (const float*)d_in[4];
  const float* embed = (const float*)d_in[5];
  const float* w3    = (const float*)d_in[6];
  const float* b3    = (const float*)d_in[7];
  const float* w4    = (const float*)d_in[8];
  const float* b4    = (const float*)d_in[9];
  const float* w5    = (const float*)d_in[10];
  const float* b5    = (const float*)d_in[11];
  const float* lin_W = (const float*)d_in[12];
  const float* lin_b = (const float*)d_in[13];

  char* ws = (char*)d_ws;
  _Float16* wbigF  = (_Float16*)(ws + OFF_WBIGF);
  _Float16* wF     = (_Float16*)(ws + OFF_WF);
  float*    txtOut = (float*)   (ws + OFF_TEXT);
  float*    hBuf   = (float*)   (ws + OFF_H);
  _Float16* hfrag  = (_Float16*)(ws + OFF_HFRAG);
  float*    asrc   = (float*)   (ws + OFF_ASRC);
  float*    adstW  = (float*)   (ws + OFF_ADST);
  float*    outP   = (float*)d_out;

  prep_weights_kernel<<<800, 256, 0, stream>>>(w3, w4, w5, W, wbigF, wF);
  textcnn_kernel<<<BN, 256, 0, stream>>>(text, embed, b3, b4, b5, wbigF,
                                         lin_W, lin_b, txtOut);
  hproj_kernel<<<BN / 128, 256, 0, stream>>>(inp, txtOut, wF, hBuf);
  hfrag_kernel<<<4096, 256, 0, stream>>>(hBuf, hfrag);
  attvec_kernel<<<BN / 256, 256, 0, stream>>>(hBuf, a_vec, asrc, adstW);
  attn_kernel<<<BB * 128, 256, 0, stream>>>(adj, asrc, adstW, hfrag, outP);
}
